// STGNN_model_83150566851113
// MI455X (gfx1250) — matrix operand order repaired and checked by reading, not compiled, not run on hardware
//
#include <hip/hip_runtime.h>

typedef __attribute__((ext_vector_type(16))) _Float16 v16h;
typedef __attribute__((ext_vector_type(8)))  _Float16 v8h;
typedef __attribute__((ext_vector_type(8)))  float    v8f;

// ---------------------------------------------------------------------------
// Optional CDNA5 async global->LDS copy path (ASYNCcnt). Guarded so that a
// toolchain without the builtins falls back to plain vector copies.
// Probe round 3 revealed the builtin prototype:
//   void __builtin_amdgcn_global_load_async_to_lds_b128(
//       int __attribute__((vector_size(16))) AS1* src,
//       int __attribute__((vector_size(16))) AS3* dst, imm offset, imm cpol)
// ---------------------------------------------------------------------------
#if defined(__AMDGCN__) && __has_builtin(__builtin_amdgcn_global_load_async_to_lds_b128) && __has_builtin(__builtin_amdgcn_s_wait_asynccnt)
#define USE_ASYNC_COPY 1
#else
#define USE_ASYNC_COPY 0
#endif

#if USE_ASYNC_COPY
typedef int b128_t __attribute__((vector_size(16)));
typedef __attribute__((address_space(1))) b128_t* gas_ptr;
typedef __attribute__((address_space(3))) b128_t* las_ptr;
// 16B global->LDS async copy. Generic->AS1 is numerically identity; per the
// ISA aperture rules the low 32 bits of a generic LDS address are the LDS
// byte offset, so the AS3 pointer is built from the truncated address.
__device__ __forceinline__ void async_copy16(const void* gsrc, void* ldst) {
  __builtin_amdgcn_global_load_async_to_lds_b128(
      (gas_ptr)(unsigned long long)gsrc,
      (las_ptr)(unsigned)(unsigned long long)ldst, 0, 0);
}
#endif

// ---------------------------------------------------------------------------
// CDNA5 WMMA 16x16x32 f16 fragment helpers (CDNA5 ISA layouts).
//
// A (16x32, MxK), row-major source: lane L holds row M = mBase + (L&15);
// its 16 halves are K = kBase + hi16 + {0..15} (hi16 = 16 for
// lanes >= 16) -> two contiguous, 16B-aligned 8-half runs = 2x ds_load_b128.
//
// B (32x16, KxN): stored pre-packed in register order so each lane reads its
// 16 halves from one contiguous 32B run = 2x ds_load_b128.
//   element (k,n) -> (((k>>5)*nTiles + (n>>4))*32 + ((k>>4)&1)*16 + (n&15))*16
//                    + (k&15)
//
// C/D (16x16 f32, 8 VGPRs): VGPR r -> M = r (lanes 0-15), M = 8+r (lanes
// 16-31); N = lane&15.
// ---------------------------------------------------------------------------
__device__ __forceinline__ int packB(int k, int n, int nTiles) {
  return ((((k >> 5) * nTiles + (n >> 4)) * 32) + (((k >> 4) & 1) * 16) +
          (n & 15)) * 16 + (k & 15);
}

__device__ __forceinline__ v16h frag_a(const _Float16* A, int mBase, int kBase,
                                       int ld, int lane) {
  const _Float16* p = A + (mBase + (lane & 15)) * ld + kBase + ((lane >> 4) << 4);
  v8h lo = *(const v8h*)p;         // K = +0..7
  v8h hi = *(const v8h*)(p + 8);   // K = +8..15
  return __builtin_shufflevector(lo, hi, 0, 1, 2, 3, 4, 5, 6, 7,
                                 8, 9, 10, 11, 12, 13, 14, 15);
}

__device__ __forceinline__ v16h frag_bp(const _Float16* Bp, int kBase, int nBase,
                                        int nTiles, int lane) {
  const _Float16* p =
      Bp + ((((kBase >> 5) * nTiles + (nBase >> 4)) * 32) + lane) * 16;
  v8h lo = *(const v8h*)p;
  v8h hi = *(const v8h*)(p + 8);
  return __builtin_shufflevector(lo, hi, 0, 1, 2, 3, 4, 5, 6, 7,
                                 8, 9, 10, 11, 12, 13, 14, 15);
}

// 16x16 tile of (M x 64)*(64 x N), K=64 in two 16x16x32 WMMA steps.
__device__ __forceinline__ v8f mm64(const _Float16* A, int mBase,
                                    const _Float16* Bp, int nBase,
                                    int lda, int nTilesB, v8f acc, int lane) {
#pragma unroll
  for (int kk = 0; kk < 64; kk += 32) {
    v16h af = frag_a(A, mBase, kk, lda, lane);
    v16h bf = frag_bp(Bp, kk, nBase, nTilesB, lane);
    acc = __builtin_amdgcn_wmma_f32_16x16x32_f16(false, af, false, bf,
                                                 (short)0, acc, false, false);
  }
  return acc;
}

// ---------------------------------------------------------------------------
// Kernel A: per-(b,l) 64x64 graph problem. Gram -> exp(-d2) -> top-8 -> Cheb.
// Output g (f16), time-major: g[(l*16384 + b*64 + n)*64 + h].
//
// 56KB LDS pool, phase-aliased:
//   [ 0K.. 8K) sXh   row-major X (A operand), live everywhere
//   [ 8K..24K) f32 X staging -> sim (f32) -> sT1 row (8K..16K) + sT1p (16K..24K)
//   [24K..32K) sXTp packed -> sSq -> sAdj row -> sT2 row
//   [32K..56K) sXp packed (to T1 phase) -> sFp packed filters (from M2 phase)
// ---------------------------------------------------------------------------
__global__ void __launch_bounds__(128)
stgnn_graphconv(const float* __restrict__ x, const float* __restrict__ filters,
                _Float16* __restrict__ g) {
  __shared__ __align__(16) unsigned char pool[57344];
  _Float16* sXh    = (_Float16*)pool;
  float*    sStage = (float*)(pool + 8192);
  float*    sSim   = (float*)(pool + 8192);
  _Float16* sT1    = (_Float16*)(pool + 8192);
  _Float16* sT1p   = (_Float16*)(pool + 16384);
  _Float16* sXTp   = (_Float16*)(pool + 24576);
  float*    sSq    = (float*)(pool + 24576);
  _Float16* sAdj   = (_Float16*)(pool + 24576);
  _Float16* sT2    = (_Float16*)(pool + 24576);
  _Float16* sXp    = (_Float16*)(pool + 32768);
  _Float16* sFp    = (_Float16*)(pool + 32768);

  const int m = blockIdx.x;
  const int b = m >> 5, l = m & 31;
  const int tid = threadIdx.x, lane = tid & 31, wave = tid >> 5;

  const float* xb = x + (size_t)b * 64 * 2048 + l * 64;

#if USE_ASYNC_COPY
  // Stage the 64x64 f32 tile into LDS with async DMA (1024 x 16B chunks).
  {
#pragma unroll
    for (int i2 = 0; i2 < 8; ++i2) {
      int i = tid + 128 * i2;
      int row = i >> 4, c4 = (i & 15) << 2;
      async_copy16(xb + (size_t)row * 2048 + c4, sStage + i * 4);
    }
    __builtin_amdgcn_s_wait_asynccnt(0);
  }
  __syncthreads();
#endif

  // Phase 0: X -> f16 row-major (A) + packed-B X and X^T copies.
  {
    int row = tid >> 1, c0 = (tid & 1) * 32;
#if USE_ASYNC_COPY
    const float* src = sStage + row * 64 + c0;
#else
    const float* src = xb + (size_t)row * 2048 + c0;
#endif
#pragma unroll
    for (int c4 = 0; c4 < 32; c4 += 4) {
      float4 v = *(const float4*)(src + c4);
      float vv[4] = {v.x, v.y, v.z, v.w};
#pragma unroll
      for (int j = 0; j < 4; ++j) {
        int c = c0 + c4 + j;
        _Float16 h = (_Float16)vv[j];
        sXh[row * 64 + c]      = h;
        sXp[packB(row, c, 4)]  = h;  // X as B:   B[k=row][n=c]
        sXTp[packB(c, row, 4)] = h;  // X^T as B: B[k=c][n=row]
      }
    }
  }
  __syncthreads();

  // Gram: G = X * X^T (4 tiles per wave)
  v8f accG[4];
#pragma unroll
  for (int i = 0; i < 4; ++i) {
    int t = wave * 4 + i;
    v8f z = {};
    accG[i] = mm64(sXh, (t >> 2) * 16, sXTp, (t & 3) * 16, 64, 4, z, lane);
  }
  __syncthreads();  // XTp dead; its region hosts sSq next

  // Row norms from the same f16 data the WMMA consumed (diag d2 ~ exactly 0).
  if (tid < 64) {
    const _Float16* r = sXh + tid * 64;
    float s = 0.f;
#pragma unroll
    for (int c = 0; c < 64; ++c) { float v = (float)r[c]; s += v * v; }
    sSq[tid] = s;
  }
  __syncthreads();

  // sim = exp(-max(sq_i + sq_j - 2G, 0))
#pragma unroll
  for (int i = 0; i < 4; ++i) {
    int t = wave * 4 + i;
    int col  = (t & 3) * 16 + (lane & 15);
    int rOff = (t >> 2) * 16 + ((lane >> 4) << 3);
#pragma unroll
    for (int r = 0; r < 8; ++r) {
      int mr = rOff + r;
      float d2 = sSq[mr] + sSq[col] - 2.f * accG[i][r];
      sSim[mr * 64 + col] = __expf(-fmaxf(d2, 0.f));
    }
  }
  __syncthreads();

  // top-8 per row -> adj (row-major A operand; overwrites sSq/XTp region)
  if (tid < 64) {
    const float* rowp = sSim + tid * 64;
    unsigned long long chosen = 0ull;
    for (int kk = 0; kk < 8; ++kk) {
      float best = -1.f; int bi = 0;
      for (int c = 0; c < 64; ++c) {
        float v = rowp[c];
        if (!((chosen >> c) & 1ull) && v > best) { best = v; bi = c; }
      }
      chosen |= (1ull << bi);
    }
    for (int c = 0; c < 64; ++c) {
      float v = ((chosen >> c) & 1ull) ? rowp[c] : 0.f;
      sAdj[tid * 64 + c] = (_Float16)v;
    }
  }
  __syncthreads();

  // T1 = adj @ X  -> row-major + packed-B copies (overwrite sim)
#pragma unroll
  for (int i = 0; i < 4; ++i) {
    int t = wave * 4 + i;
    int mB = (t >> 2) * 16, nB = (t & 3) * 16;
    v8f z = {};
    v8f a1 = mm64(sAdj, mB, sXp, nB, 64, 4, z, lane);
    int col  = nB + (lane & 15);
    int rOff = mB + ((lane >> 4) << 3);
#pragma unroll
    for (int r = 0; r < 8; ++r) {
      _Float16 h = (_Float16)a1[r];
      sT1[(rOff + r) * 64 + col]    = h;
      sT1p[packB(rOff + r, col, 4)] = h;  // T1 as B: k = node row, n = feature
    }
  }
  __syncthreads();

  // M2 = adj @ T1 (accumulators stay in registers); meanwhile pack filters
  // into the region vacated by sXp.
  v8f accM2[4];
#pragma unroll
  for (int i = 0; i < 4; ++i) {
    int t = wave * 4 + i;
    v8f z = {};
    accM2[i] = mm64(sAdj, (t >> 2) * 16, sT1p, (t & 3) * 16, 64, 4, z, lane);
  }
  for (int i = tid; i < 3 * 4096; i += 128) {
    int f = i >> 12, rem = i & 4095, p = rem >> 6, h = rem & 63;
    sFp[f * 4096 + packB(p, h, 4)] = (_Float16)filters[i];
  }
  __syncthreads();

  // T2 = 2*M2 - X  (overwrites adj)
#pragma unroll
  for (int i = 0; i < 4; ++i) {
    int t = wave * 4 + i;
    int col  = (t & 3) * 16 + (lane & 15);
    int rOff = (t >> 2) * 16 + ((lane >> 4) << 3);
#pragma unroll
    for (int r = 0; r < 8; ++r) {
      float t2 = 2.f * accM2[i][r] - (float)sXh[(rOff + r) * 64 + col];
      sT2[(rOff + r) * 64 + col] = (_Float16)t2;
    }
  }
  __syncthreads();

  // out = X@F0 + T1@F1 + T2@F2 chained through the WMMA C operand
  _Float16* gBase = g + ((size_t)l * 16384 + (size_t)b * 64) * 64;
#pragma unroll
  for (int i = 0; i < 4; ++i) {
    int t = wave * 4 + i;
    int mB = (t >> 2) * 16, nB = (t & 3) * 16;
    v8f acc = {};
    acc = mm64(sXh, mB, sFp,        nB, 64, 4, acc, lane);
    acc = mm64(sT1, mB, sFp + 4096, nB, 64, 4, acc, lane);
    acc = mm64(sT2, mB, sFp + 8192, nB, 64, 4, acc, lane);
    int col  = nB + (lane & 15);
    int rOff = mB + ((lane >> 4) << 3);
#pragma unroll
    for (int r = 0; r < 8; ++r)
      gBase[(size_t)(rOff + r) * 64 + col] = (_Float16)acc[r];
  }
}

// ---------------------------------------------------------------------------
// Kernel B: GRU over 32 steps for 32 sequences per block (512 blocks), FC dot
// product fused in registers. One partial sum per block.
// ---------------------------------------------------------------------------
__global__ void __launch_bounds__(128)
stgnn_gru(const _Float16* __restrict__ g, const float* __restrict__ w_ih,
          const float* __restrict__ w_hh, const float* __restrict__ b_ih,
          const float* __restrict__ b_hh, const float* __restrict__ fc_w,
          float* __restrict__ part) {
  __shared__ __align__(16) _Float16 sWihP[64 * 192];  // packed-B, nTiles=12
  __shared__ __align__(16) _Float16 sWhhP[64 * 192];
  __shared__ float sBih[192];
  __shared__ float sBhh[192];
  __shared__ __align__(16) _Float16 sXl[32 * 64];
  __shared__ __align__(16) _Float16 sH[2][32 * 64];   // double-buffered hidden
  __shared__ float sRed[128];

  const int tid = threadIdx.x, lane = tid & 31, wave = tid >> 5;
  const int bb = blockIdx.x;
  const int s0 = bb * 32;  // first sequence of this block

  for (int i = tid; i < 192 * 64; i += 128) {
    int n = i / 64, k = i % 64;  // w_ih is [192][64] row-major; B[k][n]=w[n][k]
    sWihP[packB(k, n, 12)] = (_Float16)w_ih[i];
    sWhhP[packB(k, n, 12)] = (_Float16)w_hh[i];
  }
  for (int i = tid; i < 192; i += 128) { sBih[i] = b_ih[i]; sBhh[i] = b_hh[i]; }
  for (int i = tid; i < 32 * 64; i += 128) sH[0][i] = (_Float16)0.f;

  int cur = 0;
  float fcpart = 0.f;

  for (int l = 0; l < 32; ++l) {
    __syncthreads();  // prev step's sXl reads + H writes complete
    const _Float16* gs = g + ((size_t)l * 16384 + s0) * 64;
#if USE_ASYNC_COPY
#pragma unroll
    for (int i2 = 0; i2 < 2; ++i2) {
      int i = tid + 128 * i2;  // 256 x 16B chunks
      async_copy16(gs + i * 8, sXl + i * 8);
    }
    __builtin_amdgcn_s_wait_asynccnt(0);
#else
#pragma unroll
    for (int i2 = 0; i2 < 2; ++i2)
      ((uint4*)sXl)[tid + 128 * i2] = ((const uint4*)gs)[tid + 128 * i2];
#endif
    if (l + 1 < 32)  // global_prefetch_b8 for next timestep's input
      __builtin_prefetch(g + ((size_t)(l + 1) * 16384 + s0) * 64 + tid * 16, 0, 0);
    __syncthreads();
    const int nxt = cur ^ 1;

#pragma unroll
    for (int ti2 = 0; ti2 < 2; ++ti2) {
      int t = wave * 2 + ti2;             // 8 tile-tasks: (ti,tj) in 2x4
      int mB = (t >> 2) * 16, nB = (t & 3) * 16;
      v8f a_ir = {}, a_iz = {}, a_in = {}, a_hr = {}, a_hz = {}, a_hn = {};
#pragma unroll
      for (int kk = 0; kk < 64; kk += 32) {
        v16h aX = frag_a(sXl,     mB, kk, 64, lane);
        v16h aH = frag_a(sH[cur], mB, kk, 64, lane);
        v16h bIr = frag_bp(sWihP, kk, nB,       12, lane);
        a_ir = __builtin_amdgcn_wmma_f32_16x16x32_f16(false, aX, false, bIr, (short)0, a_ir, false, false);
        v16h bIz = frag_bp(sWihP, kk, nB + 64,  12, lane);
        a_iz = __builtin_amdgcn_wmma_f32_16x16x32_f16(false, aX, false, bIz, (short)0, a_iz, false, false);
        v16h bIn = frag_bp(sWihP, kk, nB + 128, 12, lane);
        a_in = __builtin_amdgcn_wmma_f32_16x16x32_f16(false, aX, false, bIn, (short)0, a_in, false, false);
        v16h bHr = frag_bp(sWhhP, kk, nB,       12, lane);
        a_hr = __builtin_amdgcn_wmma_f32_16x16x32_f16(false, aH, false, bHr, (short)0, a_hr, false, false);
        v16h bHz = frag_bp(sWhhP, kk, nB + 64,  12, lane);
        a_hz = __builtin_amdgcn_wmma_f32_16x16x32_f16(false, aH, false, bHz, (short)0, a_hz, false, false);
        v16h bHn = frag_bp(sWhhP, kk, nB + 128, 12, lane);
        a_hn = __builtin_amdgcn_wmma_f32_16x16x32_f16(false, aH, false, bHn, (short)0, a_hn, false, false);
      }
      int col  = nB + (lane & 15);
      int rOff = mB + ((lane >> 4) << 3);
#pragma unroll
      for (int r = 0; r < 8; ++r) {
        int mr = rOff + r;
        float rg = a_ir[r] + sBih[col]      + a_hr[r] + sBhh[col];
        float zg = a_iz[r] + sBih[64 + col] + a_hz[r] + sBhh[64 + col];
        rg = 1.f / (1.f + __expf(-rg));
        zg = 1.f / (1.f + __expf(-zg));
        float hn = a_hn[r] + sBhh[128 + col];
        float ng = tanhf(a_in[r] + sBih[128 + col] + rg * hn);
        float hold = (float)sH[cur][mr * 64 + col];
        float hnew = (1.f - zg) * ng + zg * hold;
        sH[nxt][mr * 64 + col] = (_Float16)hnew;
        int nn = (s0 + mr) & 63;  // node index of this sequence
        fcpart += hnew * fc_w[(nn * 32 + l) * 64 + col];
      }
    }
    cur ^= 1;
  }

  sRed[tid] = fcpart;
  __syncthreads();
  if (tid == 0) {
    float s = 0.f;
    for (int i = 0; i < 128; ++i) s += sRed[i];
    part[bb] = s;
  }
}

// ---------------------------------------------------------------------------
// Kernel C: combine the two per-block partials of each batch, add fc_b.
// ---------------------------------------------------------------------------
__global__ void stgnn_final(const float* __restrict__ part,
                            const float* __restrict__ fc_b,
                            float* __restrict__ out) {
  int b = threadIdx.x;  // 256 threads
  out[b] = part[2 * b] + part[2 * b + 1] + fc_b[0];
}

extern "C" void kernel_launch(void* const* d_in, const int* in_sizes, int n_in,
                              void* d_out, int out_size, void* d_ws, size_t ws_size,
                              hipStream_t stream) {
  (void)in_sizes; (void)n_in; (void)out_size; (void)ws_size;
  const float* x       = (const float*)d_in[0];
  const float* filters = (const float*)d_in[1];
  const float* w_ih    = (const float*)d_in[2];
  const float* w_hh    = (const float*)d_in[3];
  const float* b_ih    = (const float*)d_in[4];
  const float* b_hh    = (const float*)d_in[5];
  const float* fc_w    = (const float*)d_in[6];
  const float* fc_b    = (const float*)d_in[7];
  float* out = (float*)d_out;

  // Workspace: g (f16, [32][16384][64]) then 512 FC partials (f32)
  _Float16* g    = (_Float16*)d_ws;
  float*    part = (float*)((char*)d_ws +
                            (size_t)32 * 16384 * 64 * sizeof(_Float16));

  stgnn_graphconv<<<8192, 128, 0, stream>>>(x, filters, g);
  stgnn_gru<<<512, 128, 0, stream>>>(g, w_ih, w_hh, b_ih, b_hh, fc_w, part);
  stgnn_final<<<1, 256, 0, stream>>>(part, fc_b, out);
}
